// GNN_84971632984558
// MI455X (gfx1250) — compile-verified, matrix operand order checked by reading, not run on hardware
//
#include <hip/hip_runtime.h>

// GCN forward for MI455X (gfx1250, wave32).
// - Edge scatter: L2-resident gathers + hardware f32 atomics (global_atomic_add_f32).
// - Layer-2 GEMM [100000,64]x[64,128]: v_wmma_f32_16x16x32_f16, f32 accumulate.
//   B (W2) is pre-swizzled into the WMMA fragment layout so A and B fragments
//   load as global_load_b128 (2 per fragment), zero packing VALU.
// - Self-loop term fused into WMMA epilogue.

typedef __attribute__((ext_vector_type(16))) _Float16 v16h;
typedef __attribute__((ext_vector_type(8)))  float    v8f;

#define N_NODES  100000
#define N_EDGES  1600000
#define N_GRAPHS 512

__device__ __forceinline__ void atomAddF(float* p, float v) {
  unsafeAtomicAdd(p, v);   // lowers to global_atomic_add_f32 on gfx1250
}

__global__ void zero_f32(float* __restrict__ p, int n) {
  int t = blockIdx.x * blockDim.x + threadIdx.x;
  if (t < n) p[t] = 0.0f;
}

// deg[dst] += 1 per edge (deg stored in the dinv buffer)
__global__ void deg_kernel(const int* __restrict__ dst, float* __restrict__ deg, int E) {
  int t = blockIdx.x * blockDim.x + threadIdx.x;
  if (t < E) atomAddF(&deg[dst[t]], 1.0f);
}

// dinv = rsqrt(deg + 1)   (self loop)
__global__ void dinv_kernel(float* __restrict__ dinv, int n) {
  int t = blockIdx.x * blockDim.x + threadIdx.x;
  if (t < n) dinv[t] = rsqrtf(dinv[t] + 1.0f);
}

// norm[e] = dinv[src]*dinv[dst]
__global__ void norm_kernel(const int* __restrict__ src, const int* __restrict__ dst,
                            const float* __restrict__ dinv, float* __restrict__ norm, int E) {
  int t = blockIdx.x * blockDim.x + threadIdx.x;
  if (t < E) norm[t] = dinv[src[t]] * dinv[dst[t]];
}

// xw1[i][j] = x[i]*W1[j];  agg1 = xw1*dinv[i]^2  (self-loop init)
__global__ void layer1_init(const float* __restrict__ x, const float* __restrict__ W1,
                            const float* __restrict__ dinv,
                            float* __restrict__ xw1, float* __restrict__ agg1) {
  int t = blockIdx.x * blockDim.x + threadIdx.x;   // N_NODES*64 threads
  int i = t >> 6, j = t & 63;
  float xw = x[i] * W1[j];
  xw1[t] = xw;
  float di = dinv[i];
  agg1[t] = xw * di * di;
}

// Pre-swizzle W2 (f32 [64,128]) into the WMMA B-fragment register image:
// fragment (ct, kh) = 32 lanes x 8 dwords; dword q of lane holds the f16 pair
// B[k0][c], B[k0+1][c] with k0 = kb + (q&3)*2 + (q>>2)*16 + kh*32,
// kb = (lane<16)?0:8, c = ct*16 + (lane&15).
__global__ void w2_pack(const float* __restrict__ W2, unsigned* __restrict__ w2p) {
  int t = blockIdx.x * blockDim.x + threadIdx.x;   // 8*2*32*8 = 4096 threads
  int q    = t & 7;
  int lane = (t >> 3) & 31;
  int kh   = (t >> 8) & 1;
  int ct   = t >> 9;
  int kb = (lane < 16) ? 0 : 8;
  int k0 = kb + (q & 3) * 2 + (q >> 2) * 16 + kh * 32;
  int c  = ct * 16 + (lane & 15);
  union { _Float16 h[2]; unsigned u; } pk;
  pk.h[0] = (_Float16)W2[(size_t)k0 * 128 + c];
  pk.h[1] = (_Float16)W2[(size_t)(k0 + 1) * 128 + c];
  w2p[t] = pk.u;
}

// 16 lanes per edge, float4 per lane: coalesced 256B row gathers + 64 f32 atomics/edge
__global__ void scatter1(const int* __restrict__ src, const int* __restrict__ dst,
                         const float* __restrict__ norm,
                         const float* __restrict__ xw1, float* __restrict__ agg1) {
  int t = blockIdx.x * blockDim.x + threadIdx.x;   // N_EDGES*16 threads
  int e = t >> 4, q = t & 15;
  int s = src[e], d = dst[e];
  float nm = norm[e];
  float4 v = *((const float4*)(xw1 + (size_t)s * 64) + q);
  float* o = agg1 + (size_t)d * 64 + q * 4;
  atomAddF(o + 0, v.x * nm);
  atomAddF(o + 1, v.y * nm);
  atomAddF(o + 2, v.z * nm);
  atomAddF(o + 3, v.w * nm);
}

// h1h = (f16) relu(agg1 + b1)
__global__ void h1_kernel(const float* __restrict__ agg1, const float* __restrict__ b1,
                          _Float16* __restrict__ h1h) {
  int t = blockIdx.x * blockDim.x + threadIdx.x;   // N_NODES*64 threads
  int j = t & 63;
  float v = agg1[t] + b1[j];
  h1h[t] = (_Float16)fmaxf(v, 0.0f);
}

// xw2 = h1h @ W2 via WMMA f16->f32. Block = 8 waves; wave w -> col tile w*16,
// block b -> row tile b*16. N=100000 = 6250*16 exactly, EXEC all-ones.
// A fragment: lane's dwords are two contiguous 4-dword runs -> 2x b128 loads.
// B fragment: pre-swizzled in w2p -> 2x b128 loads, no packing.
__global__ __launch_bounds__(256) void gemm2_kernel(const _Float16* __restrict__ A,  // [N,64]
                                                    const unsigned* __restrict__ w2p,
                                                    const float* __restrict__ dinv,
                                                    float* __restrict__ xw2,
                                                    float* __restrict__ agg2) {
  const int lane = threadIdx.x & 31;
  const int wave = threadIdx.x >> 5;
  const int r0 = blockIdx.x * 16;
  const int c0 = wave * 16;
  const int m  = lane & 15;
  const int kb = (lane < 16) ? 0 : 8;   // K sub-block per half-wave (ISA A/B layouts)
  const int n  = lane & 15;

  union frag { v16h v; uint4 u4[2]; };

  // A fragments: row of h1h is 128B; lane needs dwords kb/2+{0..3} and kb/2+{8..11}
  // (a0: K=0..31) and the same +16 dwords (a1: K=32..63).
  frag a0, a1;
  const char* arow = (const char*)(A + (size_t)(r0 + m) * 64);
  const int bo = kb * 2;                 // (kb/2) dwords * 4 bytes
  a0.u4[0] = *(const uint4*)(arow + bo);
  a0.u4[1] = *(const uint4*)(arow + bo + 32);
  a1.u4[0] = *(const uint4*)(arow + bo + 64);
  a1.u4[1] = *(const uint4*)(arow + bo + 96);

  // B fragments: fragment (ct=c0/16, kh) at dword offset ((ct*2+kh)*32 + lane)*8.
  frag b0, b1;
  const uint4* bp = (const uint4*)(w2p + ((size_t)((c0 >> 4) * 2) * 32 + lane) * 8);
  b0.u4[0] = bp[0];
  b0.u4[1] = bp[1];
  const uint4* bp1 = bp + 64;            // next kh fragment: 32 lanes * 8 dwords = 64 uint4
  b1.u4[0] = bp1[0];
  b1.u4[1] = bp1[1];

  v8f c = {};
  c = __builtin_amdgcn_wmma_f32_16x16x32_f16(false, a0.v, false, b0.v, (short)0, c, false, false);
  c = __builtin_amdgcn_wmma_f32_16x16x32_f16(false, a1.v, false, b1.v, (short)0, c, false, false);

  // C layout: VGPR r holds M = r + (lane<16 ? 0 : 8), N = lane&15.
#pragma unroll
  for (int r = 0; r < 8; ++r) {
    int M = r + ((lane < 16) ? 0 : 8);
    int row = r0 + M;
    size_t idx = (size_t)row * 128 + c0 + n;
    float val = c[r];
    xw2[idx] = val;
    float di = dinv[row];
    agg2[idx] = val * di * di;      // fused self-loop init
  }
}

// 32 lanes per edge, float4 per lane: 128 features
__global__ void scatter2(const int* __restrict__ src, const int* __restrict__ dst,
                         const float* __restrict__ norm,
                         const float* __restrict__ xw2, float* __restrict__ agg2) {
  int t = blockIdx.x * blockDim.x + threadIdx.x;   // N_EDGES*32 threads
  int e = t >> 5, q = t & 31;
  int s = src[e], d = dst[e];
  float nm = norm[e];
  float4 v = *((const float4*)(xw2 + (size_t)s * 128) + q);
  float* o = agg2 + (size_t)d * 128 + q * 4;
  atomAddF(o + 0, v.x * nm);
  atomAddF(o + 1, v.y * nm);
  atomAddF(o + 2, v.z * nm);
  atomAddF(o + 3, v.w * nm);
}

// h2 = relu(agg2 + b2); pool[batch[i]] += h2
__global__ void pool_kernel(const float* __restrict__ agg2, const float* __restrict__ b2,
                            const int* __restrict__ batch, float* __restrict__ pool) {
  int t = blockIdx.x * blockDim.x + threadIdx.x;   // N_NODES*128 threads
  int i = t >> 7, j = t & 127;
  float v = fmaxf(agg2[t] + b2[j], 0.0f);
  atomAddF(&pool[(size_t)batch[i] * 128 + j], v);
}

__global__ void cnt_kernel(const int* __restrict__ batch, float* __restrict__ cnt, int n) {
  int t = blockIdx.x * blockDim.x + threadIdx.x;
  if (t < n) atomAddF(&cnt[batch[t]], 1.0f);
}

// out[g] = relu(mean_pool @ Wl1 + bl1) @ Wl2 + bl2
__global__ __launch_bounds__(64) void mlp_kernel(const float* __restrict__ pool,
                                                 const float* __restrict__ cnt,
                                                 const float* __restrict__ Wl1,
                                                 const float* __restrict__ bl1,
                                                 const float* __restrict__ Wl2,
                                                 const float* __restrict__ bl2,
                                                 float* __restrict__ out) {
  __shared__ float sp[128];
  __shared__ float sh[64];
  int g = blockIdx.x, j = threadIdx.x;
  float inv = 1.0f / fmaxf(cnt[g], 1.0f);
  sp[j]      = pool[(size_t)g * 128 + j] * inv;
  sp[j + 64] = pool[(size_t)g * 128 + 64 + j] * inv;
  __syncthreads();
  float acc = bl1[j];
#pragma unroll 4
  for (int k = 0; k < 128; ++k) acc += sp[k] * Wl1[k * 64 + j];
  sh[j] = fmaxf(acc, 0.0f);
  __syncthreads();
  if (j < 4) {
    float a2 = bl2[j];
#pragma unroll 4
    for (int k = 0; k < 64; ++k) a2 += sh[k] * Wl2[k * 4 + j];
    out[g * 4 + j] = a2;
  }
}

extern "C" void kernel_launch(void* const* d_in, const int* in_sizes, int n_in,
                              void* d_out, int out_size, void* d_ws, size_t ws_size,
                              hipStream_t stream) {
  const float* x     = (const float*)d_in[0];
  const int*   ei    = (const int*)d_in[1];    // [2, E]
  const int*   batch = (const int*)d_in[2];
  const float* W1    = (const float*)d_in[3];
  const float* b1    = (const float*)d_in[4];
  const float* W2    = (const float*)d_in[5];
  const float* b2    = (const float*)d_in[6];
  const float* Wl1   = (const float*)d_in[7];
  const float* bl1   = (const float*)d_in[8];
  const float* Wl2   = (const float*)d_in[9];
  const float* bl2   = (const float*)d_in[10];
  float* out = (float*)d_out;

  const int* src = ei;
  const int* dst = ei + N_EDGES;

  // Workspace layout (256B aligned). xw1+agg1 region is reused as xw2 after h1h exists.
  char* ws = (char*)d_ws;
  size_t off = 0;
  auto alloc = [&](size_t bytes) -> void* {
    void* p = ws + off;
    off = (off + bytes + 255) & ~(size_t)255;
    return p;
  };
  float*    dinv    = (float*)alloc((size_t)N_NODES * 4);          // deg, then dinv
  float*    norm    = (float*)alloc((size_t)N_EDGES * 4);
  float*    region1 = (float*)alloc((size_t)N_NODES * 128 * 4);    // xw1|agg1 -> xw2
  float*    xw1     = region1;
  float*    agg1    = region1 + (size_t)N_NODES * 64;
  _Float16* h1h     = (_Float16*)alloc((size_t)N_NODES * 64 * 2);
  float*    agg2    = (float*)alloc((size_t)N_NODES * 128 * 4);
  unsigned* w2p     = (unsigned*)alloc((size_t)4096 * 4);          // B fragments
  float*    pool    = (float*)alloc((size_t)N_GRAPHS * 128 * 4);
  float*    cnt     = (float*)alloc((size_t)N_GRAPHS * 4);

  const int B = 256;
  zero_f32<<<(N_NODES + B - 1) / B, B, 0, stream>>>(dinv, N_NODES);
  zero_f32<<<(N_GRAPHS * 128 + B - 1) / B, B, 0, stream>>>(pool, N_GRAPHS * 128);
  zero_f32<<<(N_GRAPHS + B - 1) / B, B, 0, stream>>>(cnt, N_GRAPHS);

  deg_kernel<<<(N_EDGES + B - 1) / B, B, 0, stream>>>(dst, dinv, N_EDGES);
  dinv_kernel<<<(N_NODES + B - 1) / B, B, 0, stream>>>(dinv, N_NODES);
  norm_kernel<<<(N_EDGES + B - 1) / B, B, 0, stream>>>(src, dst, dinv, norm, N_EDGES);

  layer1_init<<<(size_t)N_NODES * 64 / B, B, 0, stream>>>(x, W1, dinv, xw1, agg1);
  w2_pack<<<4096 / B, B, 0, stream>>>(W2, w2p);

  scatter1<<<(size_t)N_EDGES * 16 / B, B, 0, stream>>>(src, dst, norm, xw1, agg1);
  h1_kernel<<<(size_t)N_NODES * 64 / B, B, 0, stream>>>(agg1, b1, h1h);

  gemm2_kernel<<<N_NODES / 16, 256, 0, stream>>>(h1h, w2p, dinv, /*xw2=*/region1, agg2);

  scatter2<<<(size_t)N_EDGES * 32 / B, B, 0, stream>>>(src, dst, norm, /*xw2=*/region1, agg2);

  pool_kernel<<<(size_t)N_NODES * 128 / B, B, 0, stream>>>(agg2, b2, batch, pool);
  cnt_kernel<<<(N_NODES + B - 1) / B, B, 0, stream>>>(batch, cnt, N_NODES);

  mlp_kernel<<<N_GRAPHS, 64, 0, stream>>>(pool, cnt, Wl1, bl1, Wl2, bl2, out);
}
